// CPFacLayer_32667521253912
// MI455X (gfx1250) — matrix operand order repaired and checked by reading, not compiled, not run on hardware
//
#include <hip/hip_runtime.h>

// ---------------------------------------------------------------------------
// CP-factorized layer on MI455X (gfx1250, wave32, WMMA).
//
// out[bv, m, n] = sum_k x[bv, m, k] * W[var(bv)][k, n]
//   m = t*64 + spatial   (M = 1024)
//   k = a*64 + c         (K = 2048)
//   n = p*64 + d         (N = 2048)
//   W[k,n] = sum_r cp0[var,a,p,r] * cp1[var,c,d,r]
//
// Kernel 0: stream-convert x (f32) -> x_bf16 in d_ws      (64 MiB)
// Kernel 1: materialize Wt[var][n][k] (bf16) in d_ws      (64 MiB)
// Kernel 2: batched GEMM, 128x256 block tile / 64x64 wave tile,
//           software-pipelined double-buffered LDS; both tiles staged with
//           global_load_async_to_lds_b128 (ASYNCcnt path);
//           compute via v_wmma_f32_16x16x32_bf16, f32 accumulate.
// ---------------------------------------------------------------------------

typedef __attribute__((ext_vector_type(16))) __bf16 v16bf;
typedef __attribute__((ext_vector_type(8)))  __bf16 v8bf;
typedef __attribute__((ext_vector_type(8)))  float  v8f;
typedef __attribute__((ext_vector_type(4)))  int    v4i;

#define AS1 __attribute__((address_space(1)))
#define AS3 __attribute__((address_space(3)))

#define M_DIM 1024
#define K_DIM 2048
#define N_DIM 2048
#define BM 128
#define BN 256
#define KB 32
#define LDP (KB + 8)   // padded LDS row: 40 bf16 = 80 B (16B-aligned stride)

// f32 -> bf16 round-to-nearest-even (used only in the streaming pre-passes)
__device__ inline unsigned short f2bf(float f) {
  unsigned u = __builtin_bit_cast(unsigned, f);
  unsigned r = u + 0x7fffu + ((u >> 16) & 1u);
  return (unsigned short)(r >> 16);
}
__device__ inline unsigned pk2bf(float a, float b) {
  return (unsigned)f2bf(a) | ((unsigned)f2bf(b) << 16);
}

// Async global -> LDS 16-byte copy (CDNA5 ASYNCcnt path).
__device__ inline void async_copy_b128(const void* g, void* lds_generic) {
#if __has_builtin(__builtin_amdgcn_global_load_async_to_lds_b128)
  __builtin_amdgcn_global_load_async_to_lds_b128(
      (AS1 v4i*)(size_t)g,
      (AS3 v4i*)(unsigned)(size_t)lds_generic, 0, 0);
#else
  unsigned l = (unsigned)(size_t)lds_generic;
  asm volatile("global_load_async_to_lds_b128 %0, %1, off"
               :: "v"(l), "v"(g) : "memory");
#endif
}

__device__ inline void async_wait0() {
#if __has_builtin(__builtin_amdgcn_s_wait_asynccnt)
  __builtin_amdgcn_s_wait_asynccnt(0);
#else
  asm volatile("s_wait_asynccnt 0x0" ::: "memory");
#endif
}

// --------------------------- Kernel 0: x -> bf16 ---------------------------
__global__ __launch_bounds__(256) void cvt_x_kernel(
    const float* __restrict__ x, __bf16* __restrict__ xb)
{
  size_t i = ((size_t)blockIdx.x * 256u + threadIdx.x) * 8u;
  const float4* p = (const float4*)(x + i);
  float4 f0 = p[0], f1 = p[1];
  uint4 o;
  o.x = pk2bf(f0.x, f0.y);
  o.y = pk2bf(f0.z, f0.w);
  o.z = pk2bf(f1.x, f1.y);
  o.w = pk2bf(f1.z, f1.w);
  *(uint4*)(xb + i) = o;
}

// --------------------------- Kernel 1: build Wt ----------------------------
// Wt[var][n][k], idx = var<<22 | n<<11 | k
__global__ __launch_bounds__(256) void build_w_kernel(
    const float* __restrict__ cp0,   // [8, 32, 32, 8]
    const float* __restrict__ cp1,   // [8, 64, 64, 8]
    __bf16* __restrict__ Wt)
{
  unsigned idx = blockIdx.x * 256u + threadIdx.x;
  unsigned k   = idx & 2047u;
  unsigned n   = (idx >> 11) & 2047u;
  unsigned var = idx >> 22;
  unsigned a = k >> 6, c = k & 63u;
  unsigned p = n >> 6, d = n & 63u;
  const float4* t0 = (const float4*)(cp0 + (((size_t)var * 32u + a) * 32u + p) * 8u);
  const float4* t1 = (const float4*)(cp1 + (((size_t)var * 64u + c) * 64u + d) * 8u);
  float4 a0 = t0[0], a1 = t0[1];
  float4 b0 = t1[0], b1 = t1[1];
  float s = a0.x * b0.x + a0.y * b0.y + a0.z * b0.z + a0.w * b0.w
          + a1.x * b1.x + a1.y * b1.y + a1.z * b1.z + a1.w * b1.w;
  Wt[idx] = __builtin_bit_cast(__bf16, f2bf(s));
}

// --------------------------- Kernel 2: WMMA GEMM ---------------------------
__global__ __launch_bounds__(256) void cp_gemm_kernel(
    const __bf16* __restrict__ xb16,     // [16, 1024(m), 2048(k)] bf16
    const __bf16* __restrict__ Wt,       // [8, 2048(n), 2048(k)] bf16
    const int*    __restrict__ var_idx,  // [16]
    float* __restrict__ out)             // [16, 1024, 2048] f32
{
  __shared__ __bf16 As[2][BM][LDP];   // A tiles, bf16, [m][k]   (20 KB)
  __shared__ __bf16 Bs[2][BN][LDP];   // B tiles, bf16, [n][k]   (40 KB)

  const int tid  = threadIdx.x;
  const int lane = tid & 31;
  const int wv   = tid >> 5;       // 8 waves
  const int wm   = wv & 1;         // 2 wave rows  (64 rows each)
  const int wn   = wv >> 1;        // 4 wave cols  (64 cols each)
  const int lm   = lane & 15;
  const int lh   = lane >> 4;

  const int bv    = blockIdx.z;
  const int var   = var_idx[bv];
  const int mbase = blockIdx.y * BM;
  const int nbase = blockIdx.x * BN;

  const __bf16* xb = xb16 + (size_t)bv  * M_DIM * K_DIM + (size_t)mbase * K_DIM;
  const __bf16* wb = Wt   + (size_t)var * N_DIM * K_DIM + (size_t)nbase * K_DIM;
  float*        ob = out  + (size_t)bv  * M_DIM * N_DIM;

  v8f zero = {};
  v8f acc[4][4];
  #pragma unroll
  for (int i = 0; i < 4; ++i)
    #pragma unroll
    for (int j = 0; j < 4; ++j) acc[i][j] = zero;

  // --- async tile staging (4 x b128 per row of 32 bf16) ---------------------
  auto loadA = [&](int buf, int kb) {   // 128 rows -> 512 b128, 2 per thread
    #pragma unroll
    for (int j = 0; j < 2; ++j) {
      int li  = j * 256 + tid;
      int row = li >> 2;
      int kc  = li & 3;
      async_copy_b128(xb + (size_t)row * K_DIM + kb + kc * 8,
                      &As[buf][row][kc * 8]);
    }
  };
  auto loadB = [&](int buf, int kb) {   // 256 rows -> 1024 b128, 4 per thread
    #pragma unroll
    for (int j = 0; j < 4; ++j) {
      int li  = j * 256 + tid;
      int row = li >> 2;
      int kc  = li & 3;
      async_copy_b128(wb + (size_t)row * K_DIM + kb + kc * 8,
                      &Bs[buf][row][kc * 8]);
    }
  };

  // --- prologue -------------------------------------------------------------
  loadA(0, 0);
  loadB(0, 0);
  async_wait0();
  __syncthreads();

  // --- main pipelined loop --------------------------------------------------
  for (int kb = 0; kb < K_DIM; kb += KB) {
    const int  cur      = (kb >> 5) & 1;
    const int  nxt      = cur ^ 1;
    const bool has_next = (kb + KB) < K_DIM;

    if (has_next) {                    // issue next tile's async traffic first
      loadA(nxt, kb + KB);
      loadB(nxt, kb + KB);
    }

    // A fragments: 16x32 bf16; lane holds M=lane&15,
    // K = half*16 + {0..15}   (half = lane>>4)
    v16bf afrag[4];
    #pragma unroll
    for (int fm = 0; fm < 4; ++fm) {
      int r  = wm * 64 + fm * 16 + lm;
      int k0 = lh * 16;
      v8bf lo = *(const v8bf*)&As[cur][r][k0];
      v8bf hi = *(const v8bf*)&As[cur][r][k0 + 8];
      afrag[fm] = __builtin_shufflevector(lo, hi,
          0, 1, 2, 3, 4, 5, 6, 7, 8, 9, 10, 11, 12, 13, 14, 15);
    }
    // B fragments: 32x16 bf16; lane holds N=lane&15, K = half*16 + {0..15}
    v16bf bfrag[4];
    #pragma unroll
    for (int fn = 0; fn < 4; ++fn) {
      int r  = wn * 64 + fn * 16 + lm;
      int k0 = lh * 16;
      v8bf lo = *(const v8bf*)&Bs[cur][r][k0];
      v8bf hi = *(const v8bf*)&Bs[cur][r][k0 + 8];
      bfrag[fn] = __builtin_shufflevector(lo, hi,
          0, 1, 2, 3, 4, 5, 6, 7, 8, 9, 10, 11, 12, 13, 14, 15);
    }
    #pragma unroll
    for (int fm = 0; fm < 4; ++fm)
      #pragma unroll
      for (int fn = 0; fn < 4; ++fn)
        acc[fm][fn] = __builtin_amdgcn_wmma_f32_16x16x32_bf16(
            false, afrag[fm], false, bfrag[fn],
            (short)0, acc[fm][fn], false, false);

    async_wait0();                     // next tiles landed in LDS
    __syncthreads();
  }

  // C/D layout: lane 0-15 -> N=lane, M=vgpr; lane 16-31 -> N=lane-16, M=8+vgpr
  #pragma unroll
  for (int fm = 0; fm < 4; ++fm) {
    int row0 = mbase + wm * 64 + fm * 16 + lh * 8;
    #pragma unroll
    for (int fn = 0; fn < 4; ++fn) {
      int col = nbase + wn * 64 + fn * 16 + lm;
      #pragma unroll
      for (int r8 = 0; r8 < 8; ++r8)
        __builtin_nontemporal_store(acc[fm][fn][r8],
                                    &ob[(size_t)(row0 + r8) * N_DIM + col]);
    }
  }
}

// ---------------------------------------------------------------------------
extern "C" void kernel_launch(void* const* d_in, const int* in_sizes, int n_in,
                              void* d_out, int out_size, void* d_ws, size_t ws_size,
                              hipStream_t stream) {
  (void)in_sizes; (void)n_in; (void)out_size; (void)ws_size;
  const float* xp   = (const float*)d_in[0];
  const float* cp0  = (const float*)d_in[1];
  const float* cp1  = (const float*)d_in[2];
  const int*   vidx = (const int*)d_in[3];
  float*  outp = (float*)d_out;

  // Workspace layout: [ Wt : 8*2048*2048 bf16 = 64 MiB ][ x_bf16 : 64 MiB ]
  __bf16* Wt = (__bf16*)d_ws;
  __bf16* Xb = Wt + (size_t)8 * 2048 * 2048;

  // Kernel 0: x (f32, 33.5M elems) -> bf16; 8 elems per thread
  cvt_x_kernel<<<dim3((16u * 1024u * 2048u) / (8u * 256u)), dim3(256), 0, stream>>>(
      xp, Xb);

  // Kernel 1: merge CP factors -> Wt[var][n][k] (bf16)
  build_w_kernel<<<dim3((8u * 2048u * 2048u) / 256u), dim3(256), 0, stream>>>(
      cp0, cp1, Wt);

  // Kernel 2: 16 batched GEMMs; grid = (N tiles, M tiles, bv)
  cp_gemm_kernel<<<dim3(N_DIM / BN, M_DIM / BM, 16), dim3(256), 0, stream>>>(
      Xb, Wt, vidx, outp);
}
